// CFHoTWrapper_1606317768732
// MI455X (gfx1250) — matrix operand order repaired and checked by reading, not compiled, not run on hardware
//
#include <hip/hip_runtime.h>
#include <hip/hip_bf16.h>

// ---------------------------------------------------------------------------
// CFHoT attention block for MI455X (gfx1250, wave32, WMMA bf16 w/ f32 accum)
// Fragment reads contiguous (ds_load_b128); all fragments of a K-step are
// preloaded into registers before the WMMA chain so LDS latency overlaps math.
// ---------------------------------------------------------------------------
#define S_LEN    2048
#define DMODEL   2048
#define NH       32
#define NKV      8
#define HD       64
#define NREP     (NH / NKV)
#define QSTRIDE  (NH * HD)    // 2048
#define KVSTRIDE (NKV * HD)   // 512
#define NEGINF   (-1e9f)
#define KSTEP    64

typedef __attribute__((ext_vector_type(16))) __bf16 v16bf;
typedef __attribute__((ext_vector_type(8)))  __bf16 v8bf;
typedef __attribute__((ext_vector_type(8)))  float  v8f;

union BF16x16 { v16bf v; __bf16 e[16]; };
union BF16x8  { v8bf  v; __bf16 e[8]; };
union F32x8   { v8f   v; float  e[8]; };

static __device__ __forceinline__ __bf16 bf(float x) { return (__bf16)x; }

// ---------------------------------------------------------------------------
// Kernel 1: per-row adapter risk  (fiber = h@Wf ; hmid = gelu([h,fiber]@W1+b1);
//           risk = hmid@W2 + b2).  One 256-thread block per sequence row.
// ---------------------------------------------------------------------------
__global__ __launch_bounds__(256)
void adapter_risk_kernel(const float* __restrict__ h, const float* __restrict__ Wf,
                         const float* __restrict__ W1, const float* __restrict__ b1,
                         const float* __restrict__ W2, const float* __restrict__ b2,
                         float* __restrict__ risk) {
  __shared__ float red[256];
  __shared__ float fib[16];
  __shared__ float hm[64];
  const int s = blockIdx.x;
  const int t = threadIdx.x;
  const float* hr = h + (size_t)s * DMODEL;

  // ---- fiber[16] ----
  {
    const int j = t & 15, seg = t >> 4;
    float sum = 0.f;
    for (int i = seg; i < DMODEL; i += 16) sum += hr[i] * Wf[i * 16 + j];
    red[t] = sum; __syncthreads();
    for (int off = 128; off >= 16; off >>= 1) {
      if (t < off) red[t] += red[t + off];
      __syncthreads();
    }
    if (t < 16) fib[t] = red[t];
    __syncthreads();
  }
  // ---- hmid[64] -> risk ----
  {
    const int c = t & 63, sg = t >> 6;
    float sum = 0.f;
    for (int i = sg; i < DMODEL; i += 4) sum += hr[i] * W1[i * 64 + c];
    for (int i = sg; i < 16; i += 4)     sum += fib[i] * W1[(DMODEL + i) * 64 + c];
    red[t] = sum; __syncthreads();
    if (t < 128) red[t] += red[t + 128];
    __syncthreads();
    if (t < 64) {
      float x = red[t] + red[t + 64] + b1[t];
      float g = 0.5f * x * (1.f + erff(x * 0.70710678118654752f));  // exact gelu
      hm[t] = g * W2[t];
    }
    __syncthreads();
    if (t < 32) hm[t] += hm[t + 32]; __syncthreads();
    if (t < 16) hm[t] += hm[t + 16]; __syncthreads();
    if (t < 8)  hm[t] += hm[t + 8];  __syncthreads();
    if (t < 4)  hm[t] += hm[t + 4];  __syncthreads();
    if (t < 2)  hm[t] += hm[t + 2];  __syncthreads();
    if (t == 0) risk[s] = hm[0] + hm[1] + b2[0];
  }
}

// ---------------------------------------------------------------------------
// Kernel 2: gate bias  gb[s] = gate_scale * (field - mean) / (std_ddof1 + 1e-6)
// ---------------------------------------------------------------------------
__global__ __launch_bounds__(256)
void gate_kernel(const float* __restrict__ risk, const float* __restrict__ gate_scale,
                 float* __restrict__ gb) {
  __shared__ float red[256];
  __shared__ float s_mean, s_inv;
  const int t = threadIdx.x;
  float sum = 0.f;
  for (int s = t; s < S_LEN; s += 256) sum += 0.005f * risk[s];   // (1-ALPHA)*risk
  red[t] = sum; __syncthreads();
  for (int off = 128; off > 0; off >>= 1) {
    if (t < off) red[t] += red[t + off];
    __syncthreads();
  }
  if (t == 0) s_mean = red[0] / (float)S_LEN;
  __syncthreads();
  const float mean = s_mean;
  float ssd = 0.f;
  for (int s = t; s < S_LEN; s += 256) {
    float d = 0.005f * risk[s] - mean;
    ssd += d * d;
  }
  red[t] = ssd; __syncthreads();
  for (int off = 128; off > 0; off >>= 1) {
    if (t < off) red[t] += red[t + off];
    __syncthreads();
  }
  if (t == 0) s_inv = 1.f / (sqrtf(red[0] / (float)(S_LEN - 1)) + 1e-6f);
  __syncthreads();
  const float gsv = gate_scale[0];
  const float inv = s_inv;
  for (int s = t; s < S_LEN; s += 256)
    gb[s] = gsv * (0.005f * risk[s] - mean) * inv;
}

// ---------------------------------------------------------------------------
// Kernel 3: bf16 WMMA GEMM  C[M,N] = A[M,K] @ B[K,N]   (fp32 in/out)
// 256 threads = 8 waves, macrotile 128x64, K-step 64.
// A staged row-major, B staged TRANSPOSED [n][k]; all 10 fragments of a
// K-step preloaded, then 8 back-to-back WMMAs.
// ---------------------------------------------------------------------------
__global__ __launch_bounds__(256)
void gemm_wmma_bf16(const float* __restrict__ A, const float* __restrict__ B,
                    float* __restrict__ C, int M, int N, int K) {
  __shared__ __align__(64) __bf16 As[128][KSTEP];    // [m][k]  16 KB
  __shared__ __align__(64) __bf16 BsT[64][KSTEP];    // [n][k]   8 KB
  const int t  = threadIdx.x;
  const int w  = t >> 5;              // wave id 0..7 -> 16-row stripe
  const int lp = t & 15;
  const int hf = (t >> 4) & 1;
  const int m0 = blockIdx.y * 128;
  const int n0 = blockIdx.x * 64;

  const int ar = t >> 1, ac = (t & 1) << 5;   // A coop: 32 contiguous floats
  const int bn = t & 63, bk = (t >> 6) << 4;  // B coop: 16 k-strided floats

  F32x8 acc[4];
#pragma unroll
  for (int i = 0; i < 4; ++i)
#pragma unroll
    for (int r = 0; r < 8; ++r) acc[i].e[r] = 0.f;

  // 32-bit offsets (all operands < 2 GB) -> saddr + voffset addressing
  int aofs = (m0 + ar) * K + ac;
  int bofs = bk * N + n0 + bn;

  for (int kb = 0; kb < K; kb += KSTEP) {
    // ---- stage A (row-major) ----
#pragma unroll
    for (int q = 0; q < 4; ++q) {
      BF16x8 pk;
#pragma unroll
      for (int i = 0; i < 8; ++i) pk.e[i] = bf(A[aofs + q * 8 + i]);
      *(v8bf*)&As[ar][ac + q * 8] = pk.v;
    }
    // ---- stage B transposed: lane-consecutive n => coalesced global loads ----
    {
      BF16x16 pb;
#pragma unroll
      for (int i = 0; i < 16; ++i) pb.e[i] = bf(B[bofs + i * N]);
      *(v16bf*)&BsT[bn][bk] = pb.v;
    }
    if (kb + KSTEP < K) {                       // gfx1250 global_prefetch_b8
      __builtin_prefetch(&A[aofs + KSTEP], 0, 1);
      __builtin_prefetch(&B[bofs + KSTEP * N], 0, 1);
    }
    aofs += KSTEP;
    bofs += KSTEP * N;
    __syncthreads();

    // ---- preload ALL fragments of this K-step ----
    BF16x16 a[2];
    BF16x16 b[2][4];
#pragma unroll
    for (int c = 0; c < 2; ++c) {
      a[c].v = *(const v16bf*)&As[w * 16 + lp][c * 32 + hf * 16];
#pragma unroll
      for (int tn = 0; tn < 4; ++tn)
        b[c][tn].v = *(const v16bf*)&BsT[tn * 16 + lp][c * 32 + hf * 16];
    }
    // ---- 8 back-to-back WMMAs (no D->A/B hazards, no per-op dscnt stall) ----
#pragma unroll
    for (int c = 0; c < 2; ++c)
#pragma unroll
      for (int tn = 0; tn < 4; ++tn)
        acc[tn].v = __builtin_amdgcn_wmma_f32_16x16x32_bf16(
            false, a[c].v, false, b[c][tn].v, (short)0, acc[tn].v, false, false);
    __syncthreads();
  }

#pragma unroll
  for (int tn = 0; tn < 4; ++tn)
#pragma unroll
    for (int r = 0; r < 8; ++r)
      C[(size_t)(m0 + w * 16 + r + hf * 8) * N + n0 + tn * 16 + lp] = acc[tn].e[r];
}

// ---------------------------------------------------------------------------
// Kernel 4: in-place RoPE. Each thread owns a (d, d+32) pair of one head-row.
// ---------------------------------------------------------------------------
__global__ __launch_bounds__(256)
void rope_inplace(float* __restrict__ x, const float* __restrict__ cosT,
                  const float* __restrict__ sinT, int nheads) {
  const int tid = blockIdx.x * blockDim.x + threadIdx.x;
  const int total = S_LEN * nheads * 32;
  if (tid >= total) return;
  const int d = tid & 31;
  const int hh = (tid >> 5) % nheads;
  const int s = tid / (nheads * 32);
  float* p = x + (size_t)s * nheads * HD + hh * HD;
  const float x1 = p[d], x2 = p[d + 32];
  const float c1 = cosT[s * HD + d],      s1 = sinT[s * HD + d];
  const float c2 = cosT[s * HD + d + 32], s2 = sinT[s * HD + d + 32];
  p[d]      = x1 * c1 - x2 * s1;   // x*cos + rotate_half(x)*sin
  p[d + 32] = x2 * c2 + x1 * s2;
}

// ---------------------------------------------------------------------------
// Kernel 5: flash attention, one wave per (head, 16-query tile).
// K staged row-major (QK^T B-frag contiguous), V staged TRANSPOSED [d][kcol].
// Fragments preloaded before each WMMA chain; causal mask + gate bias fused.
// ---------------------------------------------------------------------------
__global__ __launch_bounds__(32)
void attn_flash(const float* __restrict__ Q, const float* __restrict__ Kt,
                const float* __restrict__ V, const float* __restrict__ gb,
                float* __restrict__ O) {
  __shared__ __align__(64) __bf16 Ks[32][HD];    // [kcol][d]  4 KB
  __shared__ __align__(64) __bf16 VsT[HD][32];   // [d][kcol]  4 KB
  __shared__ __align__(64) __bf16 Ps[16][32];    // [q][kcol]  1 KB
  const int lane = threadIdx.x;
  const int lp = lane & 15, hf = lane >> 4;
  const int qt = blockIdx.x, head = blockIdx.y;
  const int kvh = head / NREP;
  const int q0 = qt * 16;

  // Q fragments (16x64 as two 16x32 A fragments), contiguous-k layout
  const float* qrow = Q + (size_t)(q0 + lp) * QSTRIDE + head * HD;
  BF16x16 qa[2];
#pragma unroll
  for (int c = 0; c < 2; ++c)
#pragma unroll
    for (int i = 0; i < 16; ++i)
      qa[c].e[i] = bf(qrow[c * 32 + hf * 16 + i]);

  F32x8 o[4];
#pragma unroll
  for (int dt = 0; dt < 4; ++dt)
#pragma unroll
    for (int r = 0; r < 8; ++r) o[dt].e[r] = 0.f;
  float mrow[8], lrow[8];
#pragma unroll
  for (int r = 0; r < 8; ++r) { mrow[r] = -3.0e38f; lrow[r] = 0.f; }

  // 32-bit offsets for staging
  int kofs = lane * KVSTRIDE + kvh * HD;          // K row for this lane
  int vofs = kvh * HD + lane;                     // V column base

  const int kend = q0 + 16;   // causal: keys < kend
  for (int kb = 0; kb < kend; kb += 32) {
    // ---- stage K row-major: one row per lane, wide LDS stores ----
#pragma unroll
    for (int q8 = 0; q8 < 8; ++q8) {
      BF16x8 pk;
#pragma unroll
      for (int i = 0; i < 8; ++i) pk.e[i] = bf(Kt[kofs + q8 * 8 + i]);
      *(v8bf*)&Ks[lane][q8 * 8] = pk.v;
    }
    // ---- stage V transposed: lane owns columns d=lane and d=lane+32 ----
    {
      BF16x8 col0[4], col1[4];
#pragma unroll
      for (int r = 0; r < 32; ++r) {
        col0[r >> 3].e[r & 7] = bf(V[vofs + r * KVSTRIDE]);
        col1[r >> 3].e[r & 7] = bf(V[vofs + r * KVSTRIDE + 32]);
      }
#pragma unroll
      for (int j = 0; j < 4; ++j) {
        *(v8bf*)&VsT[lane][j * 8]      = col0[j].v;
        *(v8bf*)&VsT[lane + 32][j * 8] = col1[j].v;
      }
    }
    kofs += 32 * KVSTRIDE;
    vofs += 32 * KVSTRIDE;
    __syncthreads();

    // ---- scores: S[16x32] = Q[16x64] @ K^T[64x32]; preload 4 B-frags ----
    BF16x16 kf[2][2];
#pragma unroll
    for (int tn = 0; tn < 2; ++tn) {
      kf[tn][0].v = *(const v16bf*)&Ks[tn * 16 + lp][hf * 16];       // d 0..31
      kf[tn][1].v = *(const v16bf*)&Ks[tn * 16 + lp][32 + hf * 16];  // d 32..63
    }
    F32x8 sf[2];
#pragma unroll
    for (int tn = 0; tn < 2; ++tn) {
      F32x8 a;
#pragma unroll
      for (int r = 0; r < 8; ++r) a.e[r] = 0.f;
      a.v = __builtin_amdgcn_wmma_f32_16x16x32_bf16(
          false, qa[0].v, false, kf[tn][0].v, (short)0, a.v, false, false);
      a.v = __builtin_amdgcn_wmma_f32_16x16x32_bf16(
          false, qa[1].v, false, kf[tn][1].v, (short)0, a.v, false, false);
      sf[tn] = a;
    }

    const int   c0i = kb + lp, c1i = kb + 16 + lp;
    const float g0 = gb[c0i],  g1 = gb[c1i];

    // ---- online softmax per C-layout row (8 rows per lane) ----
#pragma unroll
    for (int r = 0; r < 8; ++r) {
      const int qrw = q0 + r + hf * 8;
      float v0 = sf[0].e[r] * 0.125f + g0 + ((c0i <= qrw) ? 0.f : NEGINF);
      float v1 = sf[1].e[r] * 0.125f + g1 + ((c1i <= qrw) ? 0.f : NEGINF);
      float mx = fmaxf(v0, v1);
#pragma unroll
      for (int off = 1; off < 16; off <<= 1)
        mx = fmaxf(mx, __shfl_xor(mx, off, 32));   // reduce within half-wave
      const float mnew = fmaxf(mrow[r], mx);
      const float p0 = __expf(v0 - mnew);
      const float p1 = __expf(v1 - mnew);
      float rs = p0 + p1;
#pragma unroll
      for (int off = 1; off < 16; off <<= 1)
        rs += __shfl_xor(rs, off, 32);
      const float corr = __expf(mrow[r] - mnew);
      lrow[r] = lrow[r] * corr + rs;
      mrow[r] = mnew;
#pragma unroll
      for (int dt = 0; dt < 4; ++dt) o[dt].e[r] *= corr;
      Ps[r + hf * 8][lp]      = bf(p0);            // C-layout -> LDS transpose
      Ps[r + hf * 8][16 + lp] = bf(p1);
    }
    __syncthreads();

    // ---- O[16x64] += P[16x32] @ V[32x64]; preload P + 4 V-frags ----
    BF16x16 pa;
    pa.v = *(const v16bf*)&Ps[lp][hf * 16];
    BF16x16 vf[4];
#pragma unroll
    for (int dt = 0; dt < 4; ++dt)
      vf[dt].v = *(const v16bf*)&VsT[dt * 16 + lp][hf * 16];
#pragma unroll
    for (int dt = 0; dt < 4; ++dt)
      o[dt].v = __builtin_amdgcn_wmma_f32_16x16x32_bf16(
          false, pa.v, false, vf[dt].v, (short)0, o[dt].v, false, false);
    __syncthreads();
  }

#pragma unroll
  for (int dt = 0; dt < 4; ++dt)
#pragma unroll
    for (int r = 0; r < 8; ++r)
      O[(size_t)(q0 + r + hf * 8) * QSTRIDE + head * HD + dt * 16 + lp] =
          o[dt].e[r] / lrow[r];
}

// ---------------------------------------------------------------------------
extern "C" void kernel_launch(void* const* d_in, const int* in_sizes, int n_in,
                              void* d_out, int out_size, void* d_ws, size_t ws_size,
                              hipStream_t stream) {
  (void)in_sizes; (void)n_in; (void)out_size; (void)ws_size;
  const float* h    = (const float*)d_in[0];
  // d_in[1] = attention_mask: causal, recomputed exactly in-kernel (NEG = -1e9)
  const float* cosT = (const float*)d_in[2];
  const float* sinT = (const float*)d_in[3];
  const float* Wf   = (const float*)d_in[4];
  const float* W1   = (const float*)d_in[5];
  const float* b1   = (const float*)d_in[6];
  const float* W2   = (const float*)d_in[7];
  const float* b2   = (const float*)d_in[8];
  const float* gs   = (const float*)d_in[9];
  const float* Wq   = (const float*)d_in[10];
  const float* Wk   = (const float*)d_in[11];
  const float* Wv   = (const float*)d_in[12];
  const float* Wo   = (const float*)d_in[13];
  float* out = (float*)d_out;

  // workspace layout (floats): q(16MB) k(4MB) v(4MB) attn(16MB) risk gb
  float* ws   = (float*)d_ws;
  float* qbuf = ws;
  float* kbuf = qbuf + (size_t)S_LEN * QSTRIDE;
  float* vbuf = kbuf + (size_t)S_LEN * KVSTRIDE;
  float* abuf = vbuf + (size_t)S_LEN * KVSTRIDE;
  float* risk = abuf + (size_t)S_LEN * QSTRIDE;
  float* gb   = risk + S_LEN;

  adapter_risk_kernel<<<S_LEN, 256, 0, stream>>>(h, Wf, W1, b1, W2, b2, risk);
  gate_kernel<<<1, 256, 0, stream>>>(risk, gs, gb);

  gemm_wmma_bf16<<<dim3(QSTRIDE / 64, S_LEN / 128), 256, 0, stream>>>(
      h, Wq, qbuf, S_LEN, QSTRIDE, DMODEL);
  gemm_wmma_bf16<<<dim3(KVSTRIDE / 64, S_LEN / 128), 256, 0, stream>>>(
      h, Wk, kbuf, S_LEN, KVSTRIDE, DMODEL);
  gemm_wmma_bf16<<<dim3(KVSTRIDE / 64, S_LEN / 128), 256, 0, stream>>>(
      h, Wv, vbuf, S_LEN, KVSTRIDE, DMODEL);

  rope_inplace<<<(S_LEN * NH * 32 + 255) / 256, 256, 0, stream>>>(qbuf, cosT, sinT, NH);
  rope_inplace<<<(S_LEN * NKV * 32 + 255) / 256, 256, 0, stream>>>(kbuf, cosT, sinT, NKV);

  attn_flash<<<dim3(S_LEN / 16, NH), 32, 0, stream>>>(qbuf, kbuf, vbuf, gb, abuf);

  gemm_wmma_bf16<<<dim3(DMODEL / 64, S_LEN / 128), 256, 0, stream>>>(
      abuf, Wo, out, S_LEN, DMODEL, DMODEL);
}
